// cross_fuison_26319559590733
// MI455X (gfx1250) — compile-verified, matrix operand order checked by reading, not run on hardware
//
#include <hip/hip_runtime.h>

// ---------------------------------------------------------------------------
// CDNA5 (gfx1250) bf16 WMMA types
// ---------------------------------------------------------------------------
typedef __attribute__((ext_vector_type(16))) __bf16 bf16x16;
typedef __attribute__((ext_vector_type(8)))  float  floatx8;

union Frag32B { bf16x16 v; uint4 q[2]; };
union Pack16B { uint4 q; unsigned short s[8]; };

__device__ __forceinline__ unsigned short f2bf(float f) {
  // round-to-nearest-even f32 -> bf16
  unsigned int u = __float_as_uint(f);
  u += 0x7FFFu + ((u >> 16) & 1u);
  return (unsigned short)(u >> 16);
}

__device__ __forceinline__ bf16x16 load_frag16(const unsigned short* p0,
                                               const unsigned short* p1) {
  // CDNA5 16-bit A/B fragment: 2 x 16-byte chunks per lane (k = 8h..8h+7 and 16+8h..16+8h+7)
  Frag32B f;
  f.q[0] = *reinterpret_cast<const uint4*>(p0);
  f.q[1] = *reinterpret_cast<const uint4*>(p1);
  return f.v;
}

__device__ __forceinline__ floatx8 zero8() {
  floatx8 z;
#pragma unroll
  for (int e = 0; e < 8; ++e) z[e] = 0.0f;
  return z;
}

__device__ __forceinline__ floatx8 wmma_bf16(bf16x16 a, bf16x16 b, floatx8 c) {
  return __builtin_amdgcn_wmma_f32_16x16x32_bf16(false, a, false, b,
                                                 (short)0, c, false, false);
}

// gfx1250 async copy global -> LDS (16B per lane), tracked by ASYNCcnt
__device__ __forceinline__ void async_g2l_b128(unsigned lds_byte_addr,
                                               const void* gsrc) {
  unsigned long long ga = (unsigned long long)(uintptr_t)gsrc;
  asm volatile("global_load_async_to_lds_b128 %0, %1, off"
               :: "v"(lds_byte_addr), "v"(ga)
               : "memory");
}

__device__ __forceinline__ void wait_async0() {
#if __has_builtin(__builtin_amdgcn_s_wait_asynccnt)
  __builtin_amdgcn_s_wait_asynccnt(0);
#else
  asm volatile("s_wait_asynccnt 0x0" ::: "memory");
#endif
}

// ---------------------------------------------------------------------------
// f32 -> bf16 conversion, 8 elements per thread (b128 in, b128 out)
// ---------------------------------------------------------------------------
__global__ void cvt_bf16_kernel(const float* __restrict__ src,
                                unsigned short* __restrict__ dst, int n8) {
  int i = blockIdx.x * blockDim.x + threadIdx.x;
  if (i >= n8) return;
  const float4* s4 = reinterpret_cast<const float4*>(src) + 2 * (size_t)i;
  float4 a = s4[0];
  float4 b = s4[1];
  Pack16B o;
  o.s[0] = f2bf(a.x); o.s[1] = f2bf(a.y); o.s[2] = f2bf(a.z); o.s[3] = f2bf(a.w);
  o.s[4] = f2bf(b.x); o.s[5] = f2bf(b.y); o.s[6] = f2bf(b.z); o.s[7] = f2bf(b.w);
  reinterpret_cast<uint4*>(dst)[i] = o.q;
}

// ---------------------------------------------------------------------------
// Staged multi-output GEMM:
//   Cw[m][n] = sum_k A[m][k] * Bw[n][k] + biasw[n],  w = 0..NW-1
// A: M x K bf16 row-major; Bw: Nc x K bf16 row-major (X @ W^T layout).
// Block: 256 threads = 8 waves; block tile 128 x 64; wave tile 16 x 64 x NW.
// Weight slices (NW x 64 x 32) double-buffered in LDS via async copies and
// shared by all 8 waves; A-fragment double-buffered in registers.
// ---------------------------------------------------------------------------
template <int NW, int K, bool OUT_BF16>
__global__ __launch_bounds__(256) void gemm_staged_kernel(
    const unsigned short* __restrict__ A,
    const unsigned short* __restrict__ B0, const unsigned short* __restrict__ B1,
    const unsigned short* __restrict__ B2,
    const float* __restrict__ bias0, const float* __restrict__ bias1,
    const float* __restrict__ bias2,
    void* __restrict__ C0, void* __restrict__ C1, void* __restrict__ C2,
    int Nc) {
  constexpr int WSTR = 40;            // 80B rows: 16B-aligned, bank-spread
  constexpr int WMAT = 64 * WSTR;     // one weight slice (elements)
  constexpr int WBUF = NW * WMAT;     // one k-step buffer
  constexpr int STEPS = K / 32;

  __shared__ __align__(16) unsigned short Wst[2 * WBUF];

  const int lane = threadIdx.x & 31;
  const int wave = threadIdx.x >> 5;
  const int half = lane >> 4;
  const int lr = lane & 15;
  const int m0 = blockIdx.y * 128 + wave * 16;
  const int n0 = blockIdx.x * 64;

  const unsigned short* Bs[3] = {B0, B1, B2};
  const float* biases[3] = {bias0, bias1, bias2};
  void* Cs[3] = {C0, C1, C2};

  // async staging mapping: thread -> (w=it, row, 16B chunk)
  const int srow = threadIdx.x >> 2;        // 0..63
  const int sc = (threadIdx.x & 3) * 8;     // element offset 0,8,16,24

  floatx8 acc[NW][4];
#pragma unroll
  for (int w = 0; w < NW; ++w)
#pragma unroll
    for (int t = 0; t < 4; ++t) acc[w][t] = zero8();

  const unsigned short* arow = A + (size_t)(m0 + lr) * K + half * 8;

  // prologue: stage k-step 0 into buffer 0; preload first A fragment
  bf16x16 acur = load_frag16(arow, arow + 16);
#pragma unroll
  for (int it = 0; it < NW; ++it) {
    unsigned lds = (unsigned)(uintptr_t)(&Wst[it * WMAT + srow * WSTR + sc]);
    async_g2l_b128(lds, Bs[it] + (size_t)(n0 + srow) * K + sc);
  }
  wait_async0();
  __syncthreads();

  for (int ks = 0; ks < STEPS; ++ks) {
    const int cur = ks & 1;
    const int nxt = cur ^ 1;
    const bool more = (ks + 1 < STEPS);

    bf16x16 anext;
    if (more) {
      // stage next k-step while computing this one
#pragma unroll
      for (int it = 0; it < NW; ++it) {
        unsigned lds =
            (unsigned)(uintptr_t)(&Wst[nxt * WBUF + it * WMAT + srow * WSTR + sc]);
        async_g2l_b128(lds,
                       Bs[it] + (size_t)(n0 + srow) * K + (ks + 1) * 32 + sc);
      }
      const unsigned short* pa = arow + (ks + 1) * 32;
      anext = load_frag16(pa, pa + 16);
    }

    const unsigned short* wst_cur = Wst + cur * WBUF;
#pragma unroll
    for (int w = 0; w < NW; ++w) {
#pragma unroll
      for (int t = 0; t < 4; ++t) {
        const unsigned short* pb =
            wst_cur + w * WMAT + (t * 16 + lr) * WSTR + half * 8;
        bf16x16 b = load_frag16(pb, pb + 16);
        acc[w][t] = wmma_bf16(acur, b, acc[w][t]);
      }
    }

    wait_async0();
    __syncthreads();  // next buffer complete; all waves done with cur
    acur = anext;
  }

#pragma unroll
  for (int w = 0; w < NW; ++w) {
#pragma unroll
    for (int t = 0; t < 4; ++t) {
      const int col = n0 + t * 16 + lr;
      const float bb = biases[w][col];
#pragma unroll
      for (int i = 0; i < 8; ++i) {
        const int row = m0 + i + 8 * half;
        const float v = acc[w][t][i] + bb;
        if (OUT_BF16)
          reinterpret_cast<unsigned short*>(Cs[w])[(size_t)row * Nc + col] =
              f2bf(v);
        else
          reinterpret_cast<float*>(Cs[w])[(size_t)row * Nc + col] = v;
      }
    }
  }
}

// ---------------------------------------------------------------------------
// Flash attention: F = softmax(Q @ Ko^T * inv) @ V, d = 256, bf16 in/out.
// Br = 128 rows/block (8 waves x 16), Bc = 32 keys/iteration.
// K block: double-buffered LDS, filled by global_load_async_to_lds_b128.
// V block: loads for j+1 issued at top of iter j, transposed into LDS between
//          barriers (Vt single-buffered to fit 64KB static LDS).
// blockIdx.y selects branch.
// ---------------------------------------------------------------------------
__global__ __launch_bounds__(256) void flash_kernel(
    const unsigned short* __restrict__ qa, const unsigned short* __restrict__ ka,
    const unsigned short* __restrict__ va, unsigned short* __restrict__ fa,
    const unsigned short* __restrict__ qb, const unsigned short* __restrict__ kb,
    const unsigned short* __restrict__ vb, unsigned short* __restrict__ fb,
    const float* __restrict__ scale_p) {
  constexpr int D = 256;
  constexpr int BC = 32;
  constexpr int KSTR = 264;              // 528B rows: 16B-aligned, bank-spread
  constexpr int KBUF = BC * KSTR;        // one K buffer (elements)
  constexpr int VSTR = 40;               // 80B rows: 16B-aligned, bank-spread
  constexpr int PSTR = 40;
  constexpr int NJ = 8192 / BC;

  __shared__ __align__(16) unsigned short Kst[2 * KBUF];     // 33792 B
  __shared__ __align__(16) unsigned short Vt[D * VSTR];      // 20480 B
  __shared__ __align__(16) unsigned short Pb[8 * 16 * PSTR]; // 10240 B

  const bool brsel = (blockIdx.y != 0);
  const unsigned short* Q = brsel ? qb : qa;
  const unsigned short* Ko = brsel ? kb : ka;
  const unsigned short* V = brsel ? vb : va;
  unsigned short* F = brsel ? fb : fa;

  const int lane = threadIdx.x & 31;
  const int wave = threadIdx.x >> 5;
  const int half = lane >> 4;
  const int lr = lane & 15;
  const int m0 = blockIdx.x * 128 + wave * 16;

  // copy-lane mapping for staging (4 x 16B chunks per thread per block)
  const int ckey0 = threadIdx.x >> 5;       // 0..7, +8 per chunk
  const int cd0 = (threadIdx.x & 31) * 8;   // coalesced 16B along d

  float s = scale_p[0];
  s = fminf(fmaxf(s, 1.0f), 32.0f);  // clip(scale, 1, sqrt(4*256))
  const float inv = rsqrtf(s);

  float mrow[8], lrow[8];
  floatx8 Oacc[16];
#pragma unroll
  for (int i = 0; i < 8; ++i) { mrow[i] = -3.0e38f; lrow[i] = 0.0f; }
#pragma unroll
  for (int t = 0; t < 16; ++t) Oacc[t] = zero8();

  const unsigned short* qrow = Q + (size_t)(m0 + lr) * D + half * 8;
  unsigned short* prow_base = Pb + (size_t)wave * 16 * PSTR;

  uint4 vstage[4];

  // prologue: V(0) regs + async K(0) -> buf0, then commit Vt(0)
#pragma unroll
  for (int it = 0; it < 4; ++it) {
    const int key = ckey0 + 8 * it;
    vstage[it] = *reinterpret_cast<const uint4*>(V + (size_t)key * D + cd0);
  }
#pragma unroll
  for (int it = 0; it < 4; ++it) {
    const int key = ckey0 + 8 * it;
    unsigned lds = (unsigned)(uintptr_t)(&Kst[key * KSTR + cd0]);
    async_g2l_b128(lds, Ko + (size_t)key * D + cd0);
  }
#pragma unroll
  for (int it = 0; it < 4; ++it) {
    const int key = ckey0 + 8 * it;
    Pack16B u; u.q = vstage[it];
#pragma unroll
    for (int e = 0; e < 8; ++e) Vt[(cd0 + e) * VSTR + key] = u.s[e];
  }
  wait_async0();
  __syncthreads();

  for (int j = 0; j < NJ; ++j) {
    const int cur = j & 1;
    const int nxt = cur ^ 1;
    const bool more = (j + 1 < NJ);

    // --- prefetch next block: V -> regs (commit later), K -> LDS async ---
    if (more) {
      const size_t base = (size_t)(j + 1) * BC;
#pragma unroll
      for (int it = 0; it < 4; ++it) {
        const int key = ckey0 + 8 * it;
        vstage[it] =
            *reinterpret_cast<const uint4*>(V + (base + key) * D + cd0);
      }
#pragma unroll
      for (int it = 0; it < 4; ++it) {
        const int key = ckey0 + 8 * it;
        unsigned lds = (unsigned)(uintptr_t)(&Kst[nxt * KBUF + key * KSTR + cd0]);
        async_g2l_b128(lds, Ko + (base + key) * D + cd0);
      }
    }

    // --- S = Q @ Ko^T for this wave's 16 rows x 32 keys (K from LDS) ---
    const unsigned short* kst_cur = Kst + cur * KBUF;
    floatx8 sacc[2];
    sacc[0] = zero8();
    sacc[1] = zero8();
#pragma unroll
    for (int st = 0; st < D / 32; ++st) {
      const unsigned short* pq = qrow + st * 32;
      bf16x16 a = load_frag16(pq, pq + 16);
#pragma unroll
      for (int t = 0; t < 2; ++t) {
        const unsigned short* pk =
            kst_cur + (t * 16 + lr) * KSTR + st * 32 + half * 8;
        bf16x16 b = load_frag16(pk, pk + 16);
        sacc[t] = wmma_bf16(a, b, sacc[t]);
      }
    }

    // --- online softmax (row = VGPR index i + 8*half, cols across 16 lanes) ---
#pragma unroll
    for (int i = 0; i < 8; ++i) {
      float x0 = sacc[0][i] * inv;
      float x1 = sacc[1][i] * inv;
      float mx = fmaxf(x0, x1);
#pragma unroll
      for (int off = 1; off < 16; off <<= 1)
        mx = fmaxf(mx, __shfl_xor(mx, off, 32));
      const float mn = fmaxf(mrow[i], mx);
      const float p0 = __expf(x0 - mn);
      const float p1 = __expf(x1 - mn);
      float rs = p0 + p1;
#pragma unroll
      for (int off = 1; off < 16; off <<= 1) rs += __shfl_xor(rs, off, 32);
      const float corr = __expf(mrow[i] - mn);
      lrow[i] = lrow[i] * corr + rs;
      mrow[i] = mn;
#pragma unroll
      for (int t = 0; t < 16; ++t) Oacc[t][i] *= corr;
      unsigned short* pw = prow_base + (size_t)(i + 8 * half) * PSTR;
      pw[lr] = f2bf(p0);
      pw[16 + lr] = f2bf(p1);
    }

    // --- O += P @ V  (P from per-wave LDS region; V from transposed LDS) ---
    const unsigned short* pp = prow_base + (size_t)lr * PSTR + half * 8;
    bf16x16 pfrag = load_frag16(pp, pp + 16);
#pragma unroll
    for (int t = 0; t < 16; ++t) {
      const unsigned short* pv = Vt + (size_t)(t * 16 + lr) * VSTR + half * 8;
      bf16x16 b = load_frag16(pv, pv + 16);
      Oacc[t] = wmma_bf16(pfrag, b, Oacc[t]);
    }

    __syncthreads();  // all waves done reading Vt (and Kst[cur])

    // --- commit staged V(j+1) into Vt; then make everything visible ---
    if (more) {
#pragma unroll
      for (int it = 0; it < 4; ++it) {
        const int key = ckey0 + 8 * it;
        Pack16B u; u.q = vstage[it];
#pragma unroll
        for (int e = 0; e < 8; ++e) Vt[(cd0 + e) * VSTR + key] = u.s[e];
      }
    }
    wait_async0();
    __syncthreads();  // Kst[nxt] + Vt(j+1) ready
  }

  // --- normalize and store F (bf16) ---
#pragma unroll
  for (int i = 0; i < 8; ++i) {
    const float rl = 1.0f / lrow[i];
    const int row = m0 + i + 8 * half;
#pragma unroll
    for (int t = 0; t < 16; ++t) {
      F[(size_t)row * D + t * 16 + lr] = f2bf(Oacc[t][i] * rl);
    }
  }
}

// ---------------------------------------------------------------------------
// Host launcher
// ---------------------------------------------------------------------------
extern "C" void kernel_launch(void* const* d_in, const int* in_sizes, int n_in,
                              void* d_out, int out_size, void* d_ws,
                              size_t ws_size, hipStream_t stream) {
  (void)in_sizes; (void)n_in; (void)out_size; (void)ws_size;

  const float* x1 = (const float*)d_in[0];
  const float* x2 = (const float*)d_in[1];
  const float* Wq1 = (const float*)d_in[2];  const float* bq1 = (const float*)d_in[3];
  const float* Wk1 = (const float*)d_in[4];  const float* bk1 = (const float*)d_in[5];
  const float* Wv  = (const float*)d_in[6];  const float* bv  = (const float*)d_in[7];
  const float* Wq2 = (const float*)d_in[8];  const float* bq2 = (const float*)d_in[9];
  const float* Wk2 = (const float*)d_in[10]; const float* bk2 = (const float*)d_in[11];
  const float* We  = (const float*)d_in[12]; const float* be  = (const float*)d_in[13];
  const float* scale = (const float*)d_in[14];

  constexpr int N = 8192, DIN = 512, DOUT = 256;

  char* w = (char*)d_ws;
  size_t o = 0;
  auto alloc = [&](size_t bytes) -> unsigned short* {
    unsigned short* r = (unsigned short*)(w + o);
    o = (o + bytes + 255) & ~(size_t)255;
    return r;
  };
  unsigned short* x1b  = alloc((size_t)N * DIN * 2);
  unsigned short* x2b  = alloc((size_t)N * DIN * 2);
  unsigned short* wq1b = alloc((size_t)DOUT * DIN * 2);
  unsigned short* wk1b = alloc((size_t)DOUT * DIN * 2);
  unsigned short* wvb  = alloc((size_t)DOUT * DIN * 2);
  unsigned short* wq2b = alloc((size_t)DOUT * DIN * 2);
  unsigned short* wk2b = alloc((size_t)DOUT * DIN * 2);
  unsigned short* web  = alloc((size_t)DIN * DOUT * 2);
  unsigned short* q1b  = alloc((size_t)N * DOUT * 2);
  unsigned short* k1b  = alloc((size_t)N * DOUT * 2);
  unsigned short* v1b  = alloc((size_t)N * DOUT * 2);
  unsigned short* q2b  = alloc((size_t)N * DOUT * 2);
  unsigned short* k2b  = alloc((size_t)N * DOUT * 2);
  unsigned short* v2b  = alloc((size_t)N * DOUT * 2);
  unsigned short* f1b  = alloc((size_t)N * DOUT * 2);
  unsigned short* f2b  = alloc((size_t)N * DOUT * 2);

  auto cvt = [&](const float* s, unsigned short* d, int n) {
    int n8 = n / 8;
    cvt_bf16_kernel<<<(n8 + 255) / 256, 256, 0, stream>>>(s, d, n8);
  };
  cvt(x1, x1b, N * DIN);
  cvt(x2, x2b, N * DIN);
  cvt(Wq1, wq1b, DOUT * DIN);
  cvt(Wk1, wk1b, DOUT * DIN);
  cvt(Wv, wvb, DOUT * DIN);
  cvt(Wq2, wq2b, DOUT * DIN);
  cvt(Wk2, wk2b, DOUT * DIN);
  cvt(We, web, DIN * DOUT);

  // Fused projections: one pass over x per modality, 3 outputs each (bf16),
  // weights staged in LDS via async copies.
  dim3 gproj(DOUT / 64, N / 128);
  gemm_staged_kernel<3, DIN, true><<<gproj, 256, 0, stream>>>(
      x1b, wq1b, wk1b, wvb, bq1, bk1, bv, q1b, k1b, v1b, DOUT);
  gemm_staged_kernel<3, DIN, true><<<gproj, 256, 0, stream>>>(
      x2b, wq2b, wk2b, wvb, bq2, bk2, bv, q2b, k2b, v2b, DOUT);

  // Cross attention (branch 0: Q1,K2,V1 -> f1; branch 1: Q2,K1,V2 -> f2)
  flash_kernel<<<dim3(N / 128, 2), 256, 0, stream>>>(q1b, k2b, v1b, f1b,
                                                     q2b, k1b, v2b, f2b, scale);

  // Output projections: out = f @ We^T + be  (f32 out, concatenated)
  float* out1 = (float*)d_out;
  float* out2 = out1 + (size_t)N * DIN;
  dim3 gout(DIN / 64, N / 128);
  gemm_staged_kernel<1, DOUT, false><<<gout, 256, 0, stream>>>(
      f1b, web, nullptr, nullptr, be, nullptr, nullptr, out1, nullptr, nullptr,
      DIN);
  gemm_staged_kernel<1, DOUT, false><<<gout, 256, 0, stream>>>(
      f2b, web, nullptr, nullptr, be, nullptr, nullptr, out2, nullptr, nullptr,
      DIN);
}
